// VectorQuantizer_34411277976272
// MI455X (gfx1250) — matrix operand order repaired and checked by reading, compiled, not run on hardware
//
#include <hip/hip_runtime.h>

// ---------------------------------------------------------------------------
// VQ-VAE vector quantizer for MI455X (gfx1250).
// Compute bound: z@emb^T = 275 GFLOP -> v_wmma_f32_16x16x32_bf16.
// BM=128 row blocks => codebook (8MB bf16, L2-resident) re-read only 256x
// (2GB L2 traffic, balanced vs WMMA time). Codebook chunks staged by the
// Tensor Data Mover into double-buffered padded LDS rows (prefetch overlaps
// WMMA), waits via s_wait_tensorcnt. Two accumulator chains per wave to
// deepen the WMMA pipeline.
// ---------------------------------------------------------------------------

#define K_CODES 8192
#define DIM     512
#define N_VEC   32768          // 8*16*256
#define BM      128            // rows of z per block (8 waves x 16 rows)
#define BN      16             // codes per chunk
#define NCHUNK  (K_CODES / BN) // 512
#define LROW    520            // bf16 elems per padded LDS row (1040 B)

typedef __attribute__((ext_vector_type(16))) __bf16       v16bf;
typedef __attribute__((ext_vector_type(8)))  __bf16       v8bf;
typedef __attribute__((ext_vector_type(8)))  float        v8f;
typedef __attribute__((ext_vector_type(4)))  unsigned int u32x4;
typedef __attribute__((ext_vector_type(8)))  int          i32x8;
typedef __attribute__((ext_vector_type(4)))  int          i32x4;

#if defined(__has_builtin)
#  if __has_builtin(__builtin_amdgcn_tensor_load_to_lds)
#    define VQ_HAVE_TDM 1
#  endif
#endif
#ifndef VQ_HAVE_TDM
#  define VQ_HAVE_TDM 0
#endif

#if VQ_HAVE_TDM
// 2D TDM load: tile_rows x 512 bf16 rows global->LDS with 16B pad after each
// 1024B row (pad_interval=7 -> 256 DW, pad_amount=3 -> 4 DW): LDS row stride
// = LROW bf16. clang-23 6-arg builtin form.
__device__ __forceinline__ void tdm_load_rows(unsigned lds_byte_off,
                                              const __bf16* gsrc,
                                              unsigned tile_rows,
                                              unsigned tensor_rows) {
  unsigned long long ga = (unsigned long long)(size_t)gsrc;
  u32x4 g0;
  g0[0] = 1u;                                            // count=1 (valid D#)
  g0[1] = lds_byte_off;                                  // lds_addr
  g0[2] = (unsigned)(ga & 0xFFFFFFFFu);                  // global_addr lo
  g0[3] = (unsigned)((ga >> 32) & 0x01FFFFFFu)           // global_addr hi
        | (2u << 30);                                    // type=2 ("image")
  i32x8 g1;
  g1[0] = (int)((1u << 16)                               // data_size=1 (2B)
              | (1u << 20)                               // pad_enable
              | (7u << 22)                               // pad_interval: 256 DW
              | (3u << 25));                             // pad_amount: 4 DW
  g1[1] = (int)(512u << 16);                             // tensor_dim0 lo16
  g1[2] = (int)((tensor_rows & 0xFFFFu) << 16);          // dim0 hi16=0|dim1 lo16
  g1[3] = (int)(((tensor_rows >> 16) & 0xFFFFu)          // tensor_dim1 hi16
              | (512u << 16));                           // tile_dim0 = 512
  g1[4] = (int)(tile_rows & 0xFFFFu);                    // tile_dim1|tile_dim2=0
  g1[5] = 512;                                           // tensor_dim0_stride
  g1[6] = 0;
  g1[7] = 0;
  i32x4 gz4 = {0, 0, 0, 0};
  i32x8 gz8 = {0, 0, 0, 0, 0, 0, 0, 0};
  __builtin_amdgcn_tensor_load_to_lds(g0, g1, gz4, gz4, gz8, 0);
}
__device__ __forceinline__ void tdm_wait() {
#if defined(__has_builtin) && __has_builtin(__builtin_amdgcn_s_wait_tensorcnt)
  __builtin_amdgcn_s_wait_tensorcnt(0);
#else
  asm volatile("s_wait_tensorcnt 0x0" ::: "memory");
#endif
}
#else
__device__ __forceinline__ void lds_copy_rows(__bf16* dst, const __bf16* gsrc,
                                              int rows, int tid) {
  int total8 = rows * (DIM / 8);
  for (int i = tid; i < total8; i += 256) {
    int r = i >> 6;
    int c = (i & 63) * 8;
    *(v8bf*)&dst[r * LROW + c] = *(const v8bf*)&gsrc[(size_t)r * DIM + c];
  }
}
#endif

// ---------------- conversion kernels ----------------
__global__ __launch_bounds__(256) void cvt_z_kernel(const float* __restrict__ z,
                                                    __bf16* __restrict__ zb) {
  size_t i = ((size_t)blockIdx.x * 256 + threadIdx.x) * 4;
  float4 v = *(const float4*)(z + i);
  zb[i + 0] = (__bf16)v.x;
  zb[i + 1] = (__bf16)v.y;
  zb[i + 2] = (__bf16)v.z;
  zb[i + 3] = (__bf16)v.w;
}

__global__ __launch_bounds__(256) void cvt_e_kernel(const float* __restrict__ emb,
                                                    __bf16* __restrict__ eb,
                                                    float* __restrict__ esq) {
  const int k = blockIdx.x;
  const float* row = emb + (size_t)k * DIM;
  float s = 0.f;
  for (int c = threadIdx.x; c < DIM; c += 256) {
    float v = row[c];
    eb[(size_t)k * DIM + c] = (__bf16)v;
    s += v * v;
  }
  for (int off = 16; off; off >>= 1) s += __shfl_down(s, off);
  __shared__ float r8[8];
  if ((threadIdx.x & 31) == 0) r8[threadIdx.x >> 5] = s;
  __syncthreads();
  if (threadIdx.x == 0) {
    float t = 0.f;
    for (int w = 0; w < 8; ++w) t += r8[w];
    esq[k] = t;
  }
}

// ---------------- main argmin GEMM kernel ----------------
__global__ __launch_bounds__(256, 1)
void vq_argmin_kernel(const __bf16* __restrict__ zb, const __bf16* __restrict__ eb,
                      const float* __restrict__ esq, int* __restrict__ idx_out) {
  __shared__ __align__(16) __bf16 zs[BM * LROW];       // 133,120 B
  __shared__ __align__(16) __bf16 es[2 * BN * LROW];   //  33,280 B (dbl buf)

  const int tid  = threadIdx.x;
  const int wave = tid >> 5;       // 0..7 -> 16-row strip (no row overlap)
  const int lane = tid & 31;
  const int half = lane >> 4;
  const int l16  = lane & 15;
  const int row0 = blockIdx.x * BM;

  // Stage z tile (BM x 512 bf16) + first codebook chunk, then wait once.
#if VQ_HAVE_TDM
  if (wave == 0) {
    tdm_load_rows((unsigned)(size_t)&zs[0], zb + (size_t)row0 * DIM, BM, N_VEC);
    tdm_load_rows((unsigned)(size_t)&es[0], eb, BN, K_CODES);
    tdm_wait();
  }
#else
  lds_copy_rows(zs, zb + (size_t)row0 * DIM, BM, tid);
  lds_copy_rows(es, eb, BN, tid);
#endif
  __syncthreads();

  float bestS[8];
  int   bestI[8];
#pragma unroll
  for (int r = 0; r < 8; ++r) { bestS[r] = 3.4e38f; bestI[r] = 0; }

  const __bf16* arow = &zs[(wave * 16 + l16) * LROW];

  for (int i = 0; i < NCHUNK; ++i) {
    const __bf16* cur = &es[(i & 1) * (BN * LROW)];
#if VQ_HAVE_TDM
    if (wave == 0 && i + 1 < NCHUNK) {
      // Prefetch next chunk into the other buffer; overlaps WMMA below.
      tdm_load_rows((unsigned)(size_t)&es[((i + 1) & 1) * (BN * LROW)],
                    eb + (size_t)(i + 1) * BN * DIM, BN, K_CODES);
    }
#endif
    const __bf16* brow = cur + l16 * LROW;

    v8f acc0 = {};
    v8f acc1 = {};
#pragma unroll
    for (int kk = 0; kk < DIM; kk += 64) {
      // A (16x32 bf16): half 0 -> K {kk..+15}; half 1 -> K {kk+16..+31}
      v8bf a0 = *(const v8bf*)(arow + kk + half * 16);
      v8bf a1 = *(const v8bf*)(arow + kk + 8 + half * 16);
      v16bf a = __builtin_shufflevector(a0, a1, 0, 1, 2, 3, 4, 5, 6, 7,
                                        8, 9, 10, 11, 12, 13, 14, 15);
      v16bf b = *(const v16bf*)(brow + kk + half * 16);
      acc0 = __builtin_amdgcn_wmma_f32_16x16x32_bf16(false, a, false, b,
                                                     (short)0, acc0, false, false);
      v8bf c0v = *(const v8bf*)(arow + kk + 32 + half * 16);
      v8bf c1v = *(const v8bf*)(arow + kk + 40 + half * 16);
      v16bf a2 = __builtin_shufflevector(c0v, c1v, 0, 1, 2, 3, 4, 5, 6, 7,
                                         8, 9, 10, 11, 12, 13, 14, 15);
      v16bf b2 = *(const v16bf*)(brow + kk + 32 + half * 16);
      acc1 = __builtin_amdgcn_wmma_f32_16x16x32_bf16(false, a2, false, b2,
                                                     (short)0, acc1, false, false);
    }

    // score = ||e||^2 - 2 z.e  (||z||^2 constant per row, dropped)
    const int n = i * BN + l16;
    const float ev = esq[n];
#pragma unroll
    for (int r = 0; r < 8; ++r) {
      float s = __builtin_fmaf(-2.0f, acc0[r] + acc1[r], ev);
      if (s < bestS[r]) { bestS[r] = s; bestI[r] = n; }
    }

#if VQ_HAVE_TDM
    if (wave == 0) tdm_wait();     // prefetch for i+1 complete
#else
    __syncthreads();
    if (i + 1 < NCHUNK)
      lds_copy_rows(&es[((i + 1) & 1) * (BN * LROW)],
                    eb + (size_t)(i + 1) * BN * DIM, BN, tid);
#endif
    __syncthreads();               // cur fully consumed before reuse
  }

  // C layout: VGPR r -> row M = r + 8*half, col N = l16.
  // Each wave owns distinct rows: reduce across 16 lanes per half only.
#pragma unroll
  for (int r = 0; r < 8; ++r) {
    float s = bestS[r];
    int bi = bestI[r];
#pragma unroll
    for (int off = 1; off < 16; off <<= 1) {
      float s2 = __shfl_xor(s, off);
      int   i2 = __shfl_xor(bi, off);
      if (s2 < s || (s2 == s && i2 < bi)) { s = s2; bi = i2; }
    }
    if (l16 == 0) idx_out[row0 + wave * 16 + r + half * 8] = bi;
  }
}

// ---------------- gather z_q + fused loss ----------------
__global__ __launch_bounds__(256)
void vq_gather_loss_kernel(const float* __restrict__ z, const float* __restrict__ emb,
                           const int* __restrict__ idx, float* __restrict__ out) {
  const size_t base = ((size_t)blockIdx.x * 256 + threadIdx.x) * 4;
  float psum = 0.f;
  const int row  = (int)(base >> 9);
  const int col  = (int)(base & 511);
  const int code = idx[row];
  const float* e  = emb + (size_t)code * DIM + col;
  const float* zz = z + base;
#pragma unroll
  for (int j = 0; j < 4; ++j) {
    float q = e[j];
    float d = q - zz[j];
    psum += d * d;
    out[1 + base + j] = q;      // z_q_st forward value == z_q
  }
  for (int off = 16; off; off >>= 1) psum += __shfl_down(psum, off);
  __shared__ float red[8];
  if ((threadIdx.x & 31) == 0) red[threadIdx.x >> 5] = psum;
  __syncthreads();
  if (threadIdx.x == 0) {
    float t = 0.f;
    for (int w = 0; w < 8; ++w) t += red[w];
    // commitment + codebook losses are numerically identical -> factor 2
    atomicAdd(out, t * (2.0f / ((float)N_VEC * (float)DIM)));
  }
}

__global__ __launch_bounds__(256)
void vq_idx_out_kernel(const int* __restrict__ idx, float* __restrict__ out) {
  int t = blockIdx.x * 256 + threadIdx.x;
  if (t < N_VEC) out[1 + (size_t)N_VEC * DIM + t] = (float)idx[t];
}

// ---------------- host launcher ----------------
extern "C" void kernel_launch(void* const* d_in, const int* in_sizes, int n_in,
                              void* d_out, int out_size, void* d_ws, size_t ws_size,
                              hipStream_t stream) {
  (void)in_sizes; (void)n_in; (void)out_size; (void)ws_size;
  const float* z   = (const float*)d_in[0];   // [N_VEC, DIM] f32
  const float* emb = (const float*)d_in[1];   // [K_CODES, DIM] f32
  float* out = (float*)d_out;                 // [0]=loss, [1..]=z_q, tail=idx

  char* ws = (char*)d_ws;
  __bf16* zb  = (__bf16*)ws;                                            // 32 MB
  __bf16* eb  = (__bf16*)(ws + (size_t)N_VEC * DIM * 2);                //  8 MB
  float*  esq = (float*)(ws + (size_t)N_VEC * DIM * 2 + (size_t)K_CODES * DIM * 2);
  int*    idx = (int*)((char*)esq + (size_t)K_CODES * 4);

  (void)hipMemsetAsync(d_out, 0, sizeof(float), stream);  // zero loss accumulator

  cvt_z_kernel<<<(N_VEC * DIM) / (256 * 4), 256, 0, stream>>>(z, zb);
  cvt_e_kernel<<<K_CODES, 256, 0, stream>>>(emb, eb, esq);
  vq_argmin_kernel<<<N_VEC / BM, 256, 0, stream>>>(zb, eb, esq, idx);
  vq_gather_loss_kernel<<<(N_VEC * DIM) / (256 * 4), 256, 0, stream>>>(z, emb, idx, out);
  vq_idx_out_kernel<<<(N_VEC + 255) / 256, 256, 0, stream>>>(idx, out);
}
